// CrossAttention_64948495450565
// MI455X (gfx1250) — matrix operand order repaired and checked by reading, not compiled, not run on hardware
//
#include <hip/hip_runtime.h>

// ---------------------------------------------------------------------------
// Fused cross-attention for MI455X (gfx1250, wave32, WMMA 16x16x32 f16->f32).
//
// Roofline: ~13 GFLOP vs ~210 MB mandatory HBM traffic (input_a + out).
// f16 WMMA (f32 accumulate) makes this HBM-bound (~9us @ 23.3 TB/s); pure
// f32 WMMA (16x16x4) would be compute-bound. So all GEMM operands are f16,
// accumulation/softmax stay f32. input_a loads and out stores are
// non-temporal (streamed once); the small reused operands stay cached.
// ---------------------------------------------------------------------------

typedef __attribute__((ext_vector_type(16))) _Float16 v16h;
typedef __attribute__((ext_vector_type(8)))  _Float16 v8h;
typedef __attribute__((ext_vector_type(8)))  float    v8f;
typedef __attribute__((ext_vector_type(4)))  float    v4f;

#define BATCH 16
#define SEQ_A 6209
#define SEQ_B 256
#define FEAT  256
#define HID   64

// workspace layout (bytes), all offsets 256B-aligned
#define WS_WAT   0u                        // WaT [64][256]  f16 (32KB)
#define WS_WCT   (32u*1024u)               // WcT [256][64]  f16 (32KB)
#define WS_MB    (64u*1024u)               // Mb  [16][256][64]  f16 (512KB)
#define WS_MBT   (64u*1024u + 512u*1024u)  // MbT [16][64][256]  f16 (512KB)

static __device__ __forceinline__ v16h cat16(v8h lo, v8h hi) {
    union { v16h v; v8h h[2]; } u;
    u.h[0] = lo; u.h[1] = hi;
    return u.v;
}

// WMMA wrapper: D = A(16x32 f16) * B(32x16 f16) + C(16x16 f32)
static __device__ __forceinline__ v8f wmma_f16(v16h a, v16h b, v8f c) {
    return __builtin_amdgcn_wmma_f32_16x16x32_f16(
        /*neg_a=*/false, a, /*neg_b=*/false, b,
        /*c_mod=*/(short)0, c, /*reuse_a=*/false, /*reuse_b=*/false);
}

// ---------------------------------------------------------------------------
// Prep 1: f16 transposed weight copies so WMMA B-operands are contiguous
// 32B per-lane loads:  B[k][n] with n fixed per lane, k contiguous in memory.
// ---------------------------------------------------------------------------
__global__ void prep_weights(const float* __restrict__ Wa,   // [256][64]
                             const float* __restrict__ Wc,   // [64][256]
                             _Float16* __restrict__ WaT,     // [64][256]
                             _Float16* __restrict__ WcT)     // [256][64]
{
    int idx = blockIdx.x * blockDim.x + threadIdx.x;   // 0..16383
    int h = idx >> 8, f = idx & 255;                   // WaT[h][f] = Wa[f][h]
    WaT[idx] = (_Float16)Wa[f * HID + h];
    int f2 = idx >> 6, h2 = idx & 63;                  // WcT[f][h] = Wc[h][f]
    WcT[idx] = (_Float16)Wc[h2 * FEAT + f2];
}

// ---------------------------------------------------------------------------
// Prep 2: mapped_b = input_b @ Wb + bb, stored f16 both [j][h] and [h][j].
// 16*256 rows x 64 h, a 134 MFLOP afterthought.
// ---------------------------------------------------------------------------
__global__ void prep_mapped_b(const float* __restrict__ Bin,  // [16][256][256]
                              const float* __restrict__ Wb,   // [256][64]
                              const float* __restrict__ bb,   // [64]
                              _Float16* __restrict__ Mb,      // [16][256][64]
                              _Float16* __restrict__ MbT)     // [16][64][256]
{
    int b = blockIdx.y, j = blockIdx.x, h = threadIdx.x;
    const float* row = Bin + ((size_t)b * SEQ_B + j) * FEAT;
    float acc = bb[h];
    #pragma unroll 8
    for (int f = 0; f < FEAT; ++f)
        acc = fmaf(row[f], Wb[f * HID + h], acc);
    _Float16 v = (_Float16)acc;
    Mb [((size_t)b * SEQ_B + j) * HID + h]   = v;
    MbT[((size_t)b * HID   + h) * SEQ_B + j] = v;
}

// ---------------------------------------------------------------------------
// Main fused kernel. Block = 128 threads = 4 waves, each wave owns 16 rows.
// LDS is wave-private relayout scratch (C-layout accumulators -> A-layout
// fragments); no cross-wave sharing, no barriers. B-fragment loads are
// batched ahead of each WMMA group so one s_wait covers several WMMAs.
// ---------------------------------------------------------------------------
__global__ __launch_bounds__(128)
void attn_main(const float* __restrict__ A,      // input_a [16][6209][256]
               const float* __restrict__ ba,     // [64]
               const float* __restrict__ bc,     // [256]
               const _Float16* __restrict__ WaT, // [64][256]
               const _Float16* __restrict__ WcT, // [256][64]
               const _Float16* __restrict__ Mb,  // [16][256][64]
               const _Float16* __restrict__ MbT, // [16][64][256]
               float* __restrict__ out)          // [16][6209][256]
{
    __shared__ _Float16 lds_ma[4 * 16 * HID];    //  8KB
    __shared__ _Float16 lds_p [4 * 16 * SEQ_B];  // 32KB
    __shared__ _Float16 lds_o [4 * 16 * HID];    //  8KB

    const int wave = threadIdx.x >> 5;
    const int lane = threadIdx.x & 31;
    const int m    = lane & 15;     // row (A-frag) / col (B,C frag) within tile
    const int hs   = lane >> 4;     // lane-half select per ISA layouts
    const int b    = blockIdx.y;
    const int rowBase = blockIdx.x * 64 + wave * 16;

    _Float16* ma = lds_ma + wave * 16 * HID;
    _Float16* P  = lds_p  + wave * 16 * SEQ_B;
    _Float16* O  = lds_o  + wave * 16 * HID;

    int rload = rowBase + m;                 // clamp loads; stores predicated
    if (rload > SEQ_A - 1) rload = SEQ_A - 1;
    const float* Arow = A + ((size_t)b * SEQ_A + rload) * FEAT;

    const v8f vzero = {0.f, 0.f, 0.f, 0.f, 0.f, 0.f, 0.f, 0.f};

    // ---------- Phase A: mapped_a = (A @ Wa + ba) * sqrt(64) ----------
    v8f acc_ma[4];
    #pragma unroll
    for (int t = 0; t < 4; ++t) acc_ma[t] = vzero;

    #pragma unroll
    for (int kc = 0; kc < 8; ++kc) {         // K=256 in chunks of 32
        const float* ap = Arow + kc * 32;
        __builtin_prefetch(ap + 32, 0, 3);   // near (WGP) prefetch, next chunk
        // A-fragment in contiguous k order, same as B (streamed once -> NT loads):
        //   halves[0..15] = A[m][16*hs + i]
        v4f f0 = __builtin_nontemporal_load((const v4f*)(ap + hs * 16));
        v4f f1 = __builtin_nontemporal_load((const v4f*)(ap + hs * 16 + 4));
        v4f f2 = __builtin_nontemporal_load((const v4f*)(ap + hs * 16 + 8));
        v4f f3 = __builtin_nontemporal_load((const v4f*)(ap + hs * 16 + 12));
        v16h af;
        #pragma unroll
        for (int i = 0; i < 4; ++i) {
            af[i]      = (_Float16)f0[i];
            af[4 + i]  = (_Float16)f1[i];
            af[8 + i]  = (_Float16)f2[i];
            af[12 + i] = (_Float16)f3[i];
        }
        // batch all 4 B-fragments, then chain 4 WMMAs behind one wait
        v16h bf[4];
        #pragma unroll
        for (int ht = 0; ht < 4; ++ht)
            bf[ht] = *(const v16h*)(WaT + (ht * 16 + m) * FEAT + kc * 32 + hs * 16);
        #pragma unroll
        for (int ht = 0; ht < 4; ++ht)
            acc_ma[ht] = wmma_f16(af, bf[ht], acc_ma[ht]);
    }
    // bias + score scale (sqrt(HIDDEN)=8 folded into mapped_a), C-layout ->
    // LDS row-major f16. C: reg r, lane -> row r+8*hs, col ht*16+m.
    #pragma unroll
    for (int ht = 0; ht < 4; ++ht) {
        float bav = ba[ht * 16 + m];
        #pragma unroll
        for (int r = 0; r < 8; ++r)
            ma[(r + 8 * hs) * HID + ht * 16 + m] =
                (_Float16)((acc_ma[ht][r] + bav) * 8.0f);
    }

    // ---------- Phase B: scores = mapped_a @ mapped_b^T  (16 x 256) ----------
    v16h maf[2];
    #pragma unroll
    for (int kc = 0; kc < 2; ++kc) {
        v8h lo = *(const v8h*)(ma + m * HID + kc * 32 + hs * 16);
        v8h hi = *(const v8h*)(ma + m * HID + kc * 32 + hs * 16 + 8);
        maf[kc] = cat16(lo, hi);
    }
    v8f s[16];
    #pragma unroll
    for (int t = 0; t < 16; ++t) s[t] = vzero;
    #pragma unroll
    for (int jp = 0; jp < 8; ++jp) {         // 2 j-tiles per group
        v16h bf[4];
        #pragma unroll
        for (int u = 0; u < 2; ++u)
            #pragma unroll
            for (int kc = 0; kc < 2; ++kc)
                // B[k=h][n=j] = Mb[j][h]: contiguous per-lane row of Mb
                bf[u * 2 + kc] = *(const v16h*)(Mb
                    + ((size_t)b * SEQ_B + (jp * 2 + u) * 16 + m) * HID
                    + kc * 32 + hs * 16);
        #pragma unroll
        for (int u = 0; u < 2; ++u)
            #pragma unroll
            for (int kc = 0; kc < 2; ++kc)
                s[jp * 2 + u] = wmma_f16(maf[kc], bf[u * 2 + kc], s[jp * 2 + u]);
    }

    // ---------- Phase C: softmax over j (row = reg r + 8*hs across 16 lanes) --
    float rinv[8];
    #pragma unroll
    for (int r = 0; r < 8; ++r) {
        float mx = s[0][r];
        #pragma unroll
        for (int t = 1; t < 16; ++t) mx = fmaxf(mx, s[t][r]);
        #pragma unroll
        for (int d = 1; d < 16; d <<= 1) mx = fmaxf(mx, __shfl_xor(mx, d, 32));
        float sum = 0.f;
        #pragma unroll
        for (int t = 0; t < 16; ++t) {
            float p = __expf(s[t][r] - mx);
            s[t][r] = p;
            sum += p;
        }
        #pragma unroll
        for (int d = 1; d < 16; d <<= 1) sum += __shfl_xor(sum, d, 32);
        rinv[r] = 1.0f / sum;
    }
    // unnormalized P -> LDS f16 (normalize after PV)
    #pragma unroll
    for (int t = 0; t < 16; ++t)
        #pragma unroll
        for (int r = 0; r < 8; ++r)
            P[(r + 8 * hs) * SEQ_B + t * 16 + m] = (_Float16)s[t][r];

    // ---------- Phase D: out = P @ mapped_b  (16 x 64) ----------
    v8f o[4];
    #pragma unroll
    for (int t = 0; t < 4; ++t) o[t] = vzero;
    #pragma unroll
    for (int kc = 0; kc < 8; ++kc) {         // K = 256
        v8h lo = *(const v8h*)(P + m * SEQ_B + kc * 32 + hs * 16);
        v8h hi = *(const v8h*)(P + m * SEQ_B + kc * 32 + hs * 16 + 8);
        v16h af = cat16(lo, hi);
        v16h bf[4];
        #pragma unroll
        for (int ht = 0; ht < 4; ++ht)
            // B[k=j][n=h] = MbT[h][j]: contiguous per-lane row of MbT
            bf[ht] = *(const v16h*)(MbT + ((size_t)b * HID + ht * 16 + m) * SEQ_B
                                        + kc * 32 + hs * 16);
        #pragma unroll
        for (int ht = 0; ht < 4; ++ht)
            o[ht] = wmma_f16(af, bf[ht], o[ht]);
    }
    // normalize (1/rowsum) + stash f16 for final GEMM
    #pragma unroll
    for (int ht = 0; ht < 4; ++ht)
        #pragma unroll
        for (int r = 0; r < 8; ++r)
            O[(r + 8 * hs) * HID + ht * 16 + m] = (_Float16)(o[ht][r] * rinv[r]);

    // ---------- Phase E: final = O @ Wc + bc  (16 x 256) ----------
    v16h ofr[2];
    #pragma unroll
    for (int kc = 0; kc < 2; ++kc) {
        v8h lo = *(const v8h*)(O + m * HID + kc * 32 + hs * 16);
        v8h hi = *(const v8h*)(O + m * HID + kc * 32 + hs * 16 + 8);
        ofr[kc] = cat16(lo, hi);
    }
    v8f accf[16];
    #pragma unroll
    for (int t = 0; t < 16; ++t) accf[t] = vzero;
    #pragma unroll
    for (int fp = 0; fp < 8; ++fp) {         // 2 f-tiles per group
        v16h bf[4];
        #pragma unroll
        for (int u = 0; u < 2; ++u)
            #pragma unroll
            for (int kc = 0; kc < 2; ++kc)
                // B[k=h][n=f] = WcT[f][h]: contiguous per-lane row of WcT
                bf[u * 2 + kc] = *(const v16h*)(WcT
                    + ((fp * 2 + u) * 16 + m) * HID + kc * 32 + hs * 16);
        #pragma unroll
        for (int u = 0; u < 2; ++u)
            #pragma unroll
            for (int kc = 0; kc < 2; ++kc)
                accf[fp * 2 + u] =
                    wmma_f16(ofr[kc], bf[u * 2 + kc], accf[fp * 2 + u]);
    }
    // bias + predicated NT f32 store (predication AFTER all WMMAs)
    #pragma unroll
    for (int ft = 0; ft < 16; ++ft) {
        float bcv = bc[ft * 16 + m];
        #pragma unroll
        for (int r = 0; r < 8; ++r) {
            int row = rowBase + r + 8 * hs;
            if (row < SEQ_A)
                __builtin_nontemporal_store(
                    accf[ft][r] + bcv,
                    &out[((size_t)b * SEQ_A + row) * FEAT + ft * 16 + m]);
        }
    }
}

// ---------------------------------------------------------------------------
extern "C" void kernel_launch(void* const* d_in, const int* in_sizes, int n_in,
                              void* d_out, int out_size, void* d_ws, size_t ws_size,
                              hipStream_t stream)
{
    const float* input_a = (const float*)d_in[0];
    const float* input_b = (const float*)d_in[1];
    const float* Wa      = (const float*)d_in[2];
    const float* ba      = (const float*)d_in[3];
    const float* Wb      = (const float*)d_in[4];
    const float* bb      = (const float*)d_in[5];
    const float* Wc      = (const float*)d_in[6];
    const float* bc      = (const float*)d_in[7];
    float* out = (float*)d_out;

    char* ws = (char*)d_ws;
    _Float16* WaT = (_Float16*)(ws + WS_WAT);
    _Float16* WcT = (_Float16*)(ws + WS_WCT);
    _Float16* Mb  = (_Float16*)(ws + WS_MB);
    _Float16* MbT = (_Float16*)(ws + WS_MBT);

    prep_weights<<<64, 256, 0, stream>>>(Wa, Wc, WaT, WcT);
    prep_mapped_b<<<dim3(SEQ_B, BATCH), HID, 0, stream>>>(input_b, Wb, bb, Mb, MbT);
    attn_main<<<dim3((SEQ_A + 63) / 64, BATCH), 128, 0, stream>>>(
        input_a, ba, bc, WaT, WcT, Mb, MbT, out);
}
